// BaseLineModelEvent_6399501271212
// MI455X (gfx1250) — matrix operand order repaired and checked by reading, not compiled, not run on hardware
//
#include <hip/hip_runtime.h>

typedef __bf16 bf16;
typedef __bf16 v16bf __attribute__((ext_vector_type(16)));
typedef float  v8f   __attribute__((ext_vector_type(8)));

#define TPB 256

// ---------------------------------------------------------------------------
// WMMA GEMM:  out = maybe_relu(A[M,K] @ W[K,N] + bias[N])
//
// A: bf16, row stride `lda` (multiple of 32), zero-padded columns K..lda-1.
// W: bf16, PRE-PACKED in fragment order (k_pack_w): block (kt,nt), lane l's
//    16 elements contiguous at ((kt*ntiles+nt)*32 + l)*16.
// One wave computes an (MT*16) x (NT*16) output tile; fragment loads are
// unconditional 16B vector loads (global_load_b128), double-buffered so the
// next k-step's loads are in flight while the current k-step's WMMAs run.
//
// Fragment layouts used here (wave32; A/B use contiguous k, not the ISA table order):
//   A 16x32 bf16 : lane<16 -> row=lane,    elems = K k0+0..15
//                  lane>=16 -> row=lane-16, elems = K k0+16..31
//   B 32x16 bf16 : lane<16 -> col=lane,    elems = K k0+0..15
//                  lane>=16 -> col=lane-16, elems = K k0+16..31
//   C/D f32      : vgpr r : lane<16 -> (M=r, N=lane); lane>=16 -> (M=r+8, N=lane-16)
// ---------------------------------------------------------------------------
template <int MT, int NT>
__global__ __launch_bounds__(32)
void k_gemm_wmma(const bf16* __restrict__ A, int lda,
                 const bf16* __restrict__ Wp, int ntiles, int ntbase,
                 const float* __restrict__ bias,
                 float* __restrict__ outF,        // nullable
                 bf16* __restrict__ outH,         // nullable
                 int Kpad, int Nout, int ldo, int relu)
{
    const int mt0  = blockIdx.x * MT;
    const int nt0  = ntbase + blockIdx.y * NT;
    const int lane = threadIdx.x;
    const int half = lane >> 4;
    const int l15  = lane & 15;

    const bf16* Arow[MT];
#pragma unroll
    for (int m = 0; m < MT; ++m)
        Arow[m] = A + (size_t)((mt0 + m) * 16 + l15) * lda + half * 16;
    const bf16* __restrict__ Bbase = Wp + ((size_t)nt0 * 32 + lane) * 16;
    const size_t bstride = (size_t)ntiles * 512;   // elements per k-tile of packed W

    v8f acc[MT][NT];
#pragma unroll
    for (int m = 0; m < MT; ++m)
#pragma unroll
        for (int t = 0; t < NT; ++t)
            acc[m][t] = (v8f){0.f,0.f,0.f,0.f,0.f,0.f,0.f,0.f};

    struct FR { uint4 aq[MT][2]; uint4 bq[NT][2]; };

    auto loadF = [&](FR& F, int kk) {
        const int kt = kk >> 5;
        if (kk + 32 < Kpad) {
            __builtin_prefetch(Arow[0] + kk + 32, 0, 1);              // global_prefetch_b8
            __builtin_prefetch(Bbase + (size_t)(kt + 1) * bstride, 0, 1);
        }
#pragma unroll
        for (int m = 0; m < MT; ++m) {
            F.aq[m][0] = *(const uint4*)(Arow[m] + kk);
            F.aq[m][1] = *(const uint4*)(Arow[m] + kk + 8);
        }
        const bf16* bp = Bbase + (size_t)kt * bstride;
#pragma unroll
        for (int t = 0; t < NT; ++t) {
            F.bq[t][0] = *(const uint4*)(bp + t * 512);
            F.bq[t][1] = *(const uint4*)(bp + t * 512 + 8);
        }
    };
    auto computeF = [&](const FR& F) {
#pragma unroll
        for (int m = 0; m < MT; ++m) {
            union { uint4 q[2]; v16bf v; } av;
            av.q[0] = F.aq[m][0]; av.q[1] = F.aq[m][1];
#pragma unroll
            for (int t = 0; t < NT; ++t) {
                union { uint4 q[2]; v16bf v; } bv;
                bv.q[0] = F.bq[t][0]; bv.q[1] = F.bq[t][1];
                acc[m][t] = __builtin_amdgcn_wmma_f32_16x16x32_bf16(
                    false, av.v, false, bv.v, (short)0, acc[m][t], false, false);
            }
        }
    };

    // software-pipelined k loop (double-buffered fragments)
    FR F0, F1;
    loadF(F0, 0);
    int k0 = 0;
    for (;;) {
        bool more = (k0 + 32 < Kpad);
        if (more) loadF(F1, k0 + 32);
        computeF(F0);
        k0 += 32;
        if (!more) break;
        more = (k0 + 32 < Kpad);
        if (more) loadF(F0, k0 + 32);
        computeF(F1);
        k0 += 32;
        if (!more) break;
    }

#pragma unroll
    for (int m = 0; m < MT; ++m)
#pragma unroll
        for (int t = 0; t < NT; ++t) {
            const int col = (nt0 + t) * 16 + l15;
            const float bv = (col < Nout) ? bias[col] : 0.f;
#pragma unroll
            for (int r = 0; r < 8; ++r) {
                const int mm = (mt0 + m) * 16 + r + half * 8;
                float v = acc[m][t][r] + bv;
                if (relu) v = fmaxf(v, 0.f);
                if (outH) outH[(size_t)mm * ldo + col] = (bf16)v;     // padding cols get 0
                if (outF && col < Nout) outF[(size_t)mm * ldo + col] = v;
            }
        }
}

// Pack W f32[K,N] row-major -> bf16 fragment-order blocks, zero-padded.
// packed[t], t = ((kt*ntiles+nt)*32 + lane)*16 + i  maps to
//   k = kt*32 + (lane>>4)*16 + i ,  n = nt*16 + (lane&15)
__global__ void k_pack_w(const float* __restrict__ W, bf16* __restrict__ P,
                         int K, int N, int ktiles, int ntiles) {
    int t = blockIdx.x * blockDim.x + threadIdx.x;
    if (t >= ktiles * ntiles * 512) return;
    int i    = t & 15;
    int lane = (t >> 4) & 31;
    int blk  = t >> 9;
    int nt   = blk % ntiles;
    int kt   = blk / ntiles;
    int k = kt * 32 + (lane >> 4) * 16 + i;
    int n = nt * 16 + (lane & 15);
    float v = (k < K && n < N) ? W[(size_t)k * N + n] : 0.f;
    P[t] = (bf16)v;
}

// ---------------------------------------------------------------------------
// Elementwise / graph glue kernels (feature rows padded to stride 32, zeroed)
// ---------------------------------------------------------------------------
__global__ void k_zero_f32(float* __restrict__ p, int n) {
    int i = blockIdx.x * blockDim.x + threadIdx.x;
    if (i < n) p[i] = 0.f;
}

__global__ void k_node_prep(const float* __restrict__ pt, const float* __restrict__ eta,
                            const float* __restrict__ phi, const float* __restrict__ en,
                            float* __restrict__ P, int N) {
    int i = blockIdx.x * blockDim.x + threadIdx.x;
    if (i >= N) return;
    float q = pt[i], f = phi[i];
    P[i * 4 + 0] = q * cosf(f);
    P[i * 4 + 1] = q * sinf(f);
    P[i * 4 + 2] = q * sinhf(eta[i]);
    P[i * 4 + 3] = en[i];
}

__device__ __forceinline__ float safe_mass(float e0, float e1, float e2, float e3) {
    float m2 = e3 * e3 - (e0 * e0 + e1 * e1 + e2 * e2);
    return (m2 > 0.f) ? sqrtf(m2) : 0.f;
}

__global__ void k_edge_prep(const int* __restrict__ e0, const int* __restrict__ e1,
                            const float* __restrict__ eta, const float* __restrict__ phi,
                            const float* __restrict__ P,
                            bf16* __restrict__ em32, bf16* __restrict__ dr32,
                            bf16* __restrict__ f11,
                            int* __restrict__ mask, int E) {
    int e = blockIdx.x * blockDim.x + threadIdx.x;
    if (e >= E) return;
    int s = e0[e], t = e1[e];
    float xi0 = P[t * 4 + 0], xi1 = P[t * 4 + 1], xi2 = P[t * 4 + 2], xi3 = P[t * 4 + 3];
    float xj0 = P[s * 4 + 0], xj1 = P[s * 4 + 1], xj2 = P[s * 4 + 2], xj3 = P[s * 4 + 3];
    float v0 = xi0 + xj0, v1 = xi1 + xj1, v2 = xi2 + xj2, v3 = xi3 + xj3;
    float m = safe_mass(v0, v1, v2, v3);
    float de = eta[s] - eta[t], dp = phi[s] - phi[t];
    float dr = sqrtf(de * de + dp * dp);

    bf16* rm = em32 + (size_t)e * 32;
    bf16* rd = dr32 + (size_t)e * 32;
    bf16* r  = f11  + (size_t)e * 32;
#pragma unroll
    for (int j = 0; j < 32; ++j) { rm[j] = (bf16)0.f; rd[j] = (bf16)0.f; r[j] = (bf16)0.f; }
    rm[0] = (bf16)m;
    rd[0] = (bf16)dr;
    r[0] = (bf16)xi0; r[1] = (bf16)xi1; r[2] = (bf16)xi2; r[3] = (bf16)xi3;
    r[4] = (bf16)v0;  r[5] = (bf16)v1;  r[6] = (bf16)v2;  r[7] = (bf16)v3;
    r[8] = (bf16)m;   // cols 9,10 = score, filled after mlp_m
    mask[e] = 1;
}

__global__ void k_score_fill(const float* __restrict__ score, bf16* __restrict__ f11, int E) {
    int e = blockIdx.x * blockDim.x + threadIdx.x;
    if (e >= E) return;
    f11[(size_t)e * 32 + 9]  = (bf16)score[e * 2 + 0];
    f11[(size_t)e * 32 + 10] = (bf16)score[e * 2 + 1];
}

// segment_max over target nodes; complete digraph: edge (b,src=s,dst=d) has
// index b*EPE + s*(NPE-1) + (d>s ? d-1 : d)
__global__ void k_seg_max(const float* __restrict__ msg, const int* __restrict__ mask,
                          float* __restrict__ prop, int N, int NPE, int F) {
    int idx = blockIdx.x * blockDim.x + threadIdx.x;
    if (idx >= N * F) return;
    int n = idx / F, f = idx % F;
    int b = n / NPE, d = n % NPE;
    int EPE = NPE * (NPE - 1);
    float mx = -3.0e38f;
    int any = 0;
    for (int s = 0; s < NPE; ++s) {
        if (s == d) continue;
        int e = b * EPE + s * (NPE - 1) + (d > s ? d - 1 : d);
        if (mask[e]) {
            float v = msg[(size_t)e * F + f];
            mx = v > mx ? v : mx;
            any = 1;
        }
    }
    prop[idx] = any ? mx : 0.f;   // PyG fills empty max-aggregation with 0
}

__global__ void k_build26(const float* __restrict__ prop, const int* __restrict__ e0,
                          const float* __restrict__ score, const float* __restrict__ drf,
                          bf16* __restrict__ out, int E) {
    int e = blockIdx.x * blockDim.x + threadIdx.x;
    if (e >= E) return;
    int s = e0[e];
    bf16* r = out + (size_t)e * 32;
#pragma unroll
    for (int j = 0; j < 12; ++j) r[j] = (bf16)prop[s * 12 + j];
    r[12] = (bf16)score[e * 2 + 0];
    r[13] = (bf16)score[e * 2 + 1];
#pragma unroll
    for (int j = 0; j < 12; ++j) r[14 + j] = (bf16)drf[(size_t)e * 12 + j];
#pragma unroll
    for (int j = 26; j < 32; ++j) r[j] = (bf16)0.f;
}

__global__ void k_copy_f32(const float* __restrict__ s, float* __restrict__ d, int n) {
    int i = blockIdx.x * blockDim.x + threadIdx.x;
    if (i < n) d[i] = s[i];
}

__global__ void k_o_accum(const float* __restrict__ vals, const int* __restrict__ e0,
                          const int* __restrict__ mask, float* __restrict__ O, int E) {
    int e = blockIdx.x * blockDim.x + threadIdx.x;
    if (e >= E) return;
    if (mask[e]) {
        atomicAdd(&O[e0[e] * 2 + 0], vals[e * 2 + 0]);
        atomicAdd(&O[e0[e] * 2 + 1], vals[e * 2 + 1]);
    }
}

__global__ void k_mask_upd(const float* __restrict__ O, const int* __restrict__ e0,
                           int* __restrict__ mask, int E) {
    int e = blockIdx.x * blockDim.x + threadIdx.x;
    if (e >= E) return;
    int n = e0[e];
    mask[e] = mask[e] && (O[n * 2 + 1] > O[n * 2 + 0]);   // argmax==1, ties -> 0
}

__global__ void k_finalize_edges(const float* __restrict__ O, const float* __restrict__ P,
                                 const int* __restrict__ e0, const int* __restrict__ e1,
                                 float* __restrict__ actf, float* __restrict__ Pp, int E) {
    int e = blockIdx.x * blockDim.x + threadIdx.x;
    if (e >= E) return;
    float a = (O[e * 2 + 1] > O[e * 2 + 0]) ? 1.f : 0.f;
    actf[e] = a;
    if (a != 0.f) {
        int s = e0[e], t = e1[e];
        atomicAdd(&Pp[s * 4 + 0], P[t * 4 + 0]);
        atomicAdd(&Pp[s * 4 + 1], P[t * 4 + 1]);
        atomicAdd(&Pp[s * 4 + 2], P[t * 4 + 2]);
        atomicAdd(&Pp[s * 4 + 3], P[t * 4 + 3]);
    }
}

__global__ void k_tm(const float* __restrict__ Pp, float* __restrict__ tm, int N) {
    int i = blockIdx.x * blockDim.x + threadIdx.x;
    if (i >= N) return;
    tm[i] = safe_mass(Pp[i * 4 + 0], Pp[i * 4 + 1], Pp[i * 4 + 2], Pp[i * 4 + 3]);
}

__global__ void k_event(const float* __restrict__ tm, const float* __restrict__ P,
                        float* __restrict__ nT, float* __restrict__ met,
                        float* __restrict__ metp, int B, int NPE) {
    int b = blockIdx.x * blockDim.x + threadIdx.x;
    if (b >= B) return;
    float v[64];
    int n = NPE > 64 ? 64 : NPE;
    for (int k = 0; k < n; ++k) v[k] = tm[b * NPE + k];
    for (int i = 1; i < n; ++i) {                 // insertion sort ascending
        float x = v[i]; int j = i - 1;
        while (j >= 0 && v[j] > x) { v[j + 1] = v[j]; --j; }
        v[j + 1] = x;
    }
    int cnt = 0;
    for (int i = 0; i < n; ++i)
        if (v[i] > 0.f && (i == 0 || v[i] != v[i - 1])) ++cnt;
    nT[b] = (float)cnt;
    float px = 0.f, py = 0.f;
    for (int k = 0; k < NPE; ++k) {
        px += P[(b * NPE + k) * 4 + 0];
        py += P[(b * NPE + k) * 4 + 1];
    }
    met[b]  = -sqrtf(px * px + py * py);
    metp[b] = -atan2f(py, px);
}

__global__ void k_event_feats(const float* __restrict__ nT, const float* __restrict__ met,
                              const float* __restrict__ metp,
                              const float* __restrict__ Gmu, const float* __restrict__ Gmet,
                              const float* __restrict__ Gmetphi, const float* __restrict__ Gpu,
                              const float* __restrict__ Gntj,
                              bf16* __restrict__ mu_in, bf16* __restrict__ nt_in, int B) {
    int b = blockIdx.x * blockDim.x + threadIdx.x;
    if (b >= B) return;
    bf16* r = mu_in + (size_t)b * 32;
    bf16* q = nt_in + (size_t)b * 32;
#pragma unroll
    for (int j = 0; j < 32; ++j) { r[j] = (bf16)0.f; q[j] = (bf16)0.f; }
    r[0] = (bf16)nT[b];
    r[1] = (bf16)Gmu[b];
    r[2] = (bf16)Gpu[b];
    r[3] = (bf16)(met[b] - Gmet[b]);
    r[4] = (bf16)(metp[b] - Gmetphi[b]);
    r[5] = (bf16)Gntj[b];
    q[0] = (bf16)nT[b];
    q[1] = (bf16)met[b];
    q[2] = (bf16)metp[b];
    q[3] = (bf16)Gntj[b];
}

__global__ void k_topo11(const float* __restrict__ O, const float* __restrict__ actf,
                         const float* __restrict__ Pp, const int* __restrict__ e0,
                         const int* __restrict__ e1, bf16* __restrict__ out, int E) {
    int e = blockIdx.x * blockDim.x + threadIdx.x;
    if (e >= E) return;
    int s = e0[e], t = e1[e];
    bf16* r = out + (size_t)e * 32;
#pragma unroll
    for (int j = 11; j < 32; ++j) r[j] = (bf16)0.f;
    r[0] = (bf16)O[e * 2 + 0];
    r[1] = (bf16)O[e * 2 + 1];
    r[2] = (bf16)actf[e];
#pragma unroll
    for (int j = 0; j < 4; ++j) {
        r[3 + j] = (bf16)Pp[s * 4 + j];
        r[7 + j] = (bf16)Pp[t * 4 + j];
    }
}

__global__ void k_mnode5(const float* __restrict__ tm, const float* __restrict__ Pp,
                         const float* __restrict__ P, bf16* __restrict__ out, int N) {
    int i = blockIdx.x * blockDim.x + threadIdx.x;
    if (i >= N) return;
    bf16* r = out + (size_t)i * 32;
#pragma unroll
    for (int j = 5; j < 32; ++j) r[j] = (bf16)0.f;
    r[0] = (bf16)tm[i];
#pragma unroll
    for (int j = 0; j < 4; ++j) r[1 + j] = (bf16)(Pp[i * 4 + j] - P[i * 4 + j]);
}

// ---------------------------------------------------------------------------
// Host driver
// ---------------------------------------------------------------------------
extern "C" void kernel_launch(void* const* d_in, const int* in_sizes, int n_in,
                              void* d_out, int out_size, void* d_ws, size_t ws_size,
                              hipStream_t stream) {
    const int E   = in_sizes[0] / 2;
    const int B   = in_sizes[1];
    const int N   = in_sizes[2];
    const int NPE = N / B;

    const int*   edge = (const int*)d_in[0];
    const int*   e0   = edge;
    const int*   e1   = edge + E;
    const float* N_pT  = (const float*)d_in[2];
    const float* N_eta = (const float*)d_in[3];
    const float* N_phi = (const float*)d_in[4];
    const float* N_en  = (const float*)d_in[5];
    const float* G_mu      = (const float*)d_in[6];
    const float* G_met     = (const float*)d_in[7];
    const float* G_met_phi = (const float*)d_in[8];
    const float* G_pileup  = (const float*)d_in[9];
    const float* G_ntj     = (const float*)d_in[10];

    static const int WIDX[23] = {
        11, 13, 15,   // mlp        : 11->256->1024->12
        17, 19, 21,   // mlp_dr     : 1->256->1024->12
        23, 25, 27,   // mlp_m      : 1->1024->1024->2
        29, 31,       // mlp_edge   : 26->256->2
        33, 35, 37,   // mlp_topo   : 11->256->256->2
        39, 41, 43,   // mlp_mu     : 6->256->1024->1
        45, 47, 49,   // mlp_ntops  : 4->256->256->5
        51, 53, 55    // mlp_mnodetops : 5->256->256->4
    };

    char* base = (char*)d_ws;
    size_t used = 0;
    auto alloc = [&](size_t bytes) -> void* {
        size_t off = (used + 255) & ~(size_t)255;
        used = off + bytes;
        return (void*)(base + off);
    };

    // packed bf16 weights; derive (K,N) from W and bias sizes
    bf16* WB[23]; int wK[23], wN[23], wKT[23], wNT[23];
    for (int w = 0; w < 23; ++w) {
        int nW = in_sizes[WIDX[w]];
        int Nw = in_sizes[WIDX[w] + 1];          // bias length
        int Kw = nW / Nw;
        wK[w] = Kw; wN[w] = Nw;
        wKT[w] = (Kw + 31) / 32;
        wNT[w] = (Nw + 15) / 16;
        WB[w] = (bf16*)alloc((size_t)wKT[w] * wNT[w] * 512 * sizeof(bf16));
    }
    auto BI = [&](int w) -> const float* { return (const float*)d_in[WIDX[w] + 1]; };

    float* P_mu  = (float*)alloc((size_t)N * 4 * 4);
    bf16*  em32  = (bf16*) alloc((size_t)E * 32 * 2);
    bf16*  dr32  = (bf16*) alloc((size_t)E * 32 * 2);
    bf16*  f11   = (bf16*) alloc((size_t)E * 32 * 2);
    float* score = (float*)alloc((size_t)E * 2 * 4);
    float* msg   = (float*)alloc((size_t)E * 12 * 4);
    float* drf   = (float*)alloc((size_t)E * 12 * 4);
    float* prop  = (float*)alloc((size_t)N * 12 * 4);
    bf16*  e26   = (bf16*) alloc((size_t)E * 32 * 2);
    float* vals  = (float*)alloc((size_t)E * 2 * 4);
    float* Obuf  = (float*)alloc((size_t)E * 2 * 4);
    int*   mask  = (int*)  alloc((size_t)E * 4);
    float* actf  = (float*)alloc((size_t)E * 4);
    float* Pp    = (float*)alloc((size_t)N * 4 * 4);
    float* tm    = (float*)alloc((size_t)N * 4);
    float* nT    = (float*)alloc((size_t)B * 4);
    float* met   = (float*)alloc((size_t)B * 4);
    float* metp  = (float*)alloc((size_t)B * 4);
    bf16*  mu_in = (bf16*) alloc((size_t)B * 32 * 2);
    bf16*  nt_in = (bf16*) alloc((size_t)B * 32 * 2);
    bf16*  t11   = (bf16*) alloc((size_t)E * 32 * 2);
    bf16*  m5    = (bf16*) alloc((size_t)N * 32 * 2);

    // remaining workspace -> two ping-pong hidden buffers [CH,1024] bf16
    size_t rem = (ws_size > used + 512) ? (ws_size - used - 512) : 0;
    long long chll = (long long)(rem / 2 / (1024 * sizeof(bf16)));
    int CH = (int)(chll > 8192 ? 8192 : chll);
    CH &= ~31;                                   // multiple of 32 -> wide MT=2 path
    if (CH < 32) CH = 32;
    bf16* bufX = (bf16*)alloc((size_t)CH * 1024 * sizeof(bf16));
    bf16* bufY = (bf16*)alloc((size_t)CH * 1024 * sizeof(bf16));

    auto blocks = [](int n) { return (n + TPB - 1) / TPB; };

    // A: lda-padded bf16; w: weight id; oF/oH: f32 / bf16 output (ldo = Nout)
    auto gemm = [&](const bf16* A, int lda, int w, float* oF, bf16* oH,
                    int M, int relu) {
        const int Kpad = wKT[w] * 32;
        const int Nout = wN[w];
        const int nt   = wNT[w];
        const int ntg  = nt / 4;                 // groups of 4 n-tiles
        const int ntr  = nt % 4;                 // leftover tiles (NT=1 each)
        if ((M % 32) == 0) {
            if (ntg) k_gemm_wmma<2, 4><<<dim3(M / 32, ntg), 32, 0, stream>>>(
                         A, lda, WB[w], nt, 0, BI(w), oF, oH, Kpad, Nout, Nout, relu);
            if (ntr) k_gemm_wmma<2, 1><<<dim3(M / 32, ntr), 32, 0, stream>>>(
                         A, lda, WB[w], nt, ntg * 4, BI(w), oF, oH, Kpad, Nout, Nout, relu);
        } else {
            if (ntg) k_gemm_wmma<1, 4><<<dim3(M / 16, ntg), 32, 0, stream>>>(
                         A, lda, WB[w], nt, 0, BI(w), oF, oH, Kpad, Nout, Nout, relu);
            if (ntr) k_gemm_wmma<1, 1><<<dim3(M / 16, ntr), 32, 0, stream>>>(
                         A, lda, WB[w], nt, ntg * 4, BI(w), oF, oH, Kpad, Nout, Nout, relu);
        }
    };

    // 0) weight packing f32 -> fragment-ordered bf16
    for (int w = 0; w < 23; ++w) {
        int n = wKT[w] * wNT[w] * 512;
        k_pack_w<<<blocks(n), TPB, 0, stream>>>((const float*)d_in[WIDX[w]], WB[w],
                                                wK[w], wN[w], wKT[w], wNT[w]);
    }

    // 1) node + edge prep
    k_node_prep<<<blocks(N), TPB, 0, stream>>>(N_pT, N_eta, N_phi, N_en, P_mu, N);
    k_edge_prep<<<blocks(E), TPB, 0, stream>>>(e0, e1, N_eta, N_phi, P_mu,
                                               em32, dr32, f11, mask, E);

    // 2) mlp_m : em[E,1] -> 1024 -> 1024 -> score[E,2]
    for (int r0 = 0; r0 < E; r0 += CH) {
        int m = E - r0 < CH ? E - r0 : CH;
        gemm(em32 + (size_t)r0 * 32, 32,   6, nullptr, bufX, m, 1);
        gemm(bufX,              1024,      7, nullptr, bufY, m, 1);
        gemm(bufY,              1024,      8, score + (size_t)r0 * 2, nullptr, m, 0);
    }
    k_score_fill<<<blocks(E), TPB, 0, stream>>>(score, f11, E);

    // 3) mlp : f11[E,11] -> 256 -> 1024 -> msg[E,12]
    for (int r0 = 0; r0 < E; r0 += CH) {
        int m = E - r0 < CH ? E - r0 : CH;
        gemm(f11 + (size_t)r0 * 32, 32,    0, nullptr, bufX, m, 1);
        gemm(bufX,              256,       1, nullptr, bufY, m, 1);
        gemm(bufY,              1024,      2, msg + (size_t)r0 * 12, nullptr, m, 0);
    }

    // 4) mlp_dr : dr[E,1] -> 256 -> 1024 -> dr_feat[E,12]
    for (int r0 = 0; r0 < E; r0 += CH) {
        int m = E - r0 < CH ? E - r0 : CH;
        gemm(dr32 + (size_t)r0 * 32, 32,   3, nullptr, bufX, m, 1);
        gemm(bufX,              256,       4, nullptr, bufY, m, 1);
        gemm(bufY,              1024,      5, drf + (size_t)r0 * 12, nullptr, m, 0);
    }

    // 5) 3-step recursion
    for (int it = 0; it < 3; ++it) {
        k_seg_max<<<blocks(N * 12), TPB, 0, stream>>>(msg, mask, prop, N, NPE, 12);
        k_build26<<<blocks(E), TPB, 0, stream>>>(prop, e0, score, drf, e26, E);
        for (int r0 = 0; r0 < E; r0 += CH) {
            int m = E - r0 < CH ? E - r0 : CH;
            gemm(e26 + (size_t)r0 * 32, 32,  9, nullptr, bufX, m, 1);
            gemm(bufX,              256,    10, vals + (size_t)r0 * 2, nullptr, m, 0);
        }
        if (it == 0)
            k_copy_f32<<<blocks(E * 2), TPB, 0, stream>>>(vals, Obuf, E * 2);
        else
            k_o_accum<<<blocks(E), TPB, 0, stream>>>(vals, e0, mask, Obuf, E);
        k_mask_upd<<<blocks(E), TPB, 0, stream>>>(Obuf, e0, mask, E);
    }

    // 6) actf, P_mu_p scatter, masses, per-event stats
    k_zero_f32<<<blocks(N * 4), TPB, 0, stream>>>(Pp, N * 4);
    k_finalize_edges<<<blocks(E), TPB, 0, stream>>>(Obuf, P_mu, e0, e1, actf, Pp, E);
    k_tm<<<blocks(N), TPB, 0, stream>>>(Pp, tm, N);
    k_event<<<blocks(B), TPB, 0, stream>>>(tm, P_mu, nT, met, metp, B, NPE);
    k_event_feats<<<blocks(B), TPB, 0, stream>>>(nT, met, metp, G_mu, G_met, G_met_phi,
                                                 G_pileup, G_ntj, mu_in, nt_in, B);

    // 7) output heads: d_out = O_Topo[E,2] | O_mu[B,1] | O_nTops[B,5] | O_Index[N,4]
    float* dout    = (float*)d_out;
    float* o_topo  = dout;
    float* o_mu    = dout + (size_t)E * 2;
    float* o_ntops = o_mu + B;
    float* o_index = o_ntops + (size_t)B * 5;

    k_topo11<<<blocks(E), TPB, 0, stream>>>(Obuf, actf, Pp, e0, e1, t11, E);
    for (int r0 = 0; r0 < E; r0 += CH) {
        int m = E - r0 < CH ? E - r0 : CH;
        gemm(t11 + (size_t)r0 * 32, 32,   11, nullptr, bufX, m, 1);
        gemm(bufX,              256,      12, nullptr, bufY, m, 1);
        gemm(bufY,              256,      13, o_topo + (size_t)r0 * 2, nullptr, m, 0);
    }

    gemm(mu_in, 32,   14, nullptr, bufX, B, 1);
    gemm(bufX,  256,  15, nullptr, bufY, B, 1);
    gemm(bufY,  1024, 16, o_mu, nullptr, B, 0);

    gemm(nt_in, 32,   17, nullptr, bufX, B, 1);
    gemm(bufX,  256,  18, nullptr, bufY, B, 1);
    gemm(bufY,  256,  19, o_ntops, nullptr, B, 0);

    k_mnode5<<<blocks(N), TPB, 0, stream>>>(tm, Pp, P_mu, m5, N);
    gemm(m5,   32,  20, nullptr, bufX, N, 1);
    gemm(bufX, 256, 21, nullptr, bufY, N, 1);
    gemm(bufY, 256, 22, o_index, nullptr, N, 0);
}
